// MotionDecoderLayer_22814866276629
// MI455X (gfx1250) — compile-verified, matrix operand order checked by reading, not run on hardware
//
#include <hip/hip_runtime.h>
#include <hip/hip_bf16.h>
#include <math.h>

typedef __bf16 bf16;
typedef __attribute__((ext_vector_type(16))) __bf16 v16bf;
typedef __attribute__((ext_vector_type(8)))  float  v8f;

#define NB 2
#define TT 1024
#define DD 1024
#define HH 16
#define DH 64
#define EE 8
#define FF 4096
#define NTOK (NB*TT)

union FragA { v16bf v; unsigned u[8]; };
union PairU { unsigned u; bf16 h[2]; };

// ---------------------------------------------------------------------------
// LayerNorm: fp32 in -> bf16 out. One block (256 thr) per token.
// ---------------------------------------------------------------------------
__global__ __launch_bounds__(256)
void ln_kernel(const float* __restrict__ X, const float* __restrict__ g,
               const float* __restrict__ bta, bf16* __restrict__ Hout)
{
  const int n = blockIdx.x;
  const float* x = X + (size_t)n * DD;
  const int t = threadIdx.x, lane = t & 31, w = t >> 5;
  float s = 0.f, s2 = 0.f;
  for (int d = t; d < DD; d += 256) { float v = x[d]; s += v; s2 += v * v; }
  for (int d = 1; d < 32; d <<= 1) { s += __shfl_xor(s, d, 32); s2 += __shfl_xor(s2, d, 32); }
  __shared__ float rs[8], rs2[8], stats[2];
  if (lane == 0) { rs[w] = s; rs2[w] = s2; }
  __syncthreads();
  if (t == 0) {
    float a = 0.f, b2 = 0.f;
    for (int i = 0; i < 8; ++i) { a += rs[i]; b2 += rs2[i]; }
    float mu = a / (float)DD;
    stats[0] = mu;
    stats[1] = b2 / (float)DD - mu * mu;
  }
  __syncthreads();
  const float mu = stats[0];
  const float inv = rsqrtf(stats[1] + 1e-5f);
  for (int d = t; d < DD; d += 256) {
    float v = (x[d] - mu) * inv * g[d] + bta[d];
    Hout[(size_t)n * DD + d] = (bf16)v;
  }
}

// ---------------------------------------------------------------------------
// WMMA bf16 GEMM: C(MxN) = A(MxK bf16) * B(KxN fp32->bf16) + epilogue.
// Block 256 thr / 8 waves in 2x4 grid; block tile M=64 N=128, K-step 32.
// Each wave: 2 A-frags x 2 B-frags -> 4 accumulators -> 4 WMMA per step.
// A tile staged with GLOBAL_LOAD_ASYNC_TO_LDS_B128 (CDNA5 async path);
// B tile converted fp32->bf16 in flight and stored transposed [n][k].
// EPI: 0 store bf16   1 resid+bias f32   2 exact-GELU bf16   3 out += cw*(acc+bias)
// EPI 2/3 skip whole block when every token in the 64-row M-tile has cw==0.
// ---------------------------------------------------------------------------
template<int EPI>
__global__ __launch_bounds__(256)
void gemm_bf16_kernel(const bf16* __restrict__ A, const float* __restrict__ Bw,
                      const float* __restrict__ bias, const float* __restrict__ resid,
                      const float* __restrict__ cw,          // combine-weight column (stride EE)
                      bf16* __restrict__ outB, float* __restrict__ outF,
                      int M, int Nn, int K)
{
  const int m0 = blockIdx.y * 64;
  const int n0 = blockIdx.x * 128;
  const int t = threadIdx.x;
  const int lane = t & 31;
  const int wave = t >> 5;
  const int wr = wave >> 2;           // 0..1  (M direction)
  const int wc = wave & 3;            // 0..3  (N direction)

  __shared__ __align__(16) bf16 sA[64 * 32];
  __shared__ __align__(16) bf16 sBT[128 * 32];
  __shared__ int sActive;

  if (EPI == 2 || EPI == 3) {
    if (t == 0) sActive = 0;
    __syncthreads();
    if (t < 64 && cw[(size_t)(m0 + t) * EE] != 0.0f) sActive = 1;
    __syncthreads();
    if (!sActive) return;             // tile unrouted for this expert
  }

  v8f acc00, acc01, acc10, acc11;
  #pragma unroll
  for (int r = 0; r < 8; ++r) { acc00[r] = 0.f; acc01[r] = 0.f; acc10[r] = 0.f; acc11[r] = 0.f; }

  const int mfrag = lane & 15;
  const int nlane = lane & 15;
  const int kbase = (lane >> 4) << 3;

  // per-thread async A-copy coordinates: one b128 (8 halves) per K-step
  const int am = t >> 2;              // 0..63
  const int ak = (t & 3) << 3;        // 0, 8, 16, 24

  for (int k0 = 0; k0 < K; k0 += 32) {
    __syncthreads();
    // ---- stage A tile 64x32 via async global->LDS b128 ----
    {
      unsigned ldsoff = (unsigned)(size_t)&sA[am * 32 + ak];
      unsigned long long ga =
          (unsigned long long)(size_t)&A[(size_t)(m0 + am) * K + k0 + ak];
      asm volatile("global_load_async_to_lds_b128 %0, %1, off"
                   :: "v"(ldsoff), "v"(ga) : "memory");
    }
    // ---- stage B tile 32x128 fp32 -> bf16, transposed to [n][k] ----
    #pragma unroll
    for (int i = 0; i < 16; ++i) {
      int e2 = t + i * 256;
      int kk = e2 >> 7;
      int n = e2 & 127;
      float f = Bw[(size_t)(k0 + kk) * Nn + n0 + n];
      sBT[n * 32 + kk] = (bf16)f;
    }
    asm volatile("s_wait_asynccnt 0x0" ::: "memory");
    __syncthreads();

    FragA fb0, fb1, fa0, fa1;
    #pragma unroll
    for (int p = 0; p < 4; ++p) {
      const int nrow0 = (wc * 32 + nlane) * 32;
      const int nrow1 = (wc * 32 + 16 + nlane) * 32;
      fb0.u[p]     = *(const unsigned*)&sBT[nrow0 + kbase + 2 * p];
      fb0.u[4 + p] = *(const unsigned*)&sBT[nrow0 + 16 + kbase + 2 * p];
      fb1.u[p]     = *(const unsigned*)&sBT[nrow1 + kbase + 2 * p];
      fb1.u[4 + p] = *(const unsigned*)&sBT[nrow1 + 16 + kbase + 2 * p];
    }
    #pragma unroll
    for (int p = 0; p < 4; ++p) {
      const int mrow0 = (wr * 32 + mfrag) * 32;
      const int mrow1 = (wr * 32 + 16 + mfrag) * 32;
      fa0.u[p]     = *(const unsigned*)&sA[mrow0 + kbase + 2 * p];
      fa0.u[4 + p] = *(const unsigned*)&sA[mrow0 + 16 + kbase + 2 * p];
      fa1.u[p]     = *(const unsigned*)&sA[mrow1 + kbase + 2 * p];
      fa1.u[4 + p] = *(const unsigned*)&sA[mrow1 + 16 + kbase + 2 * p];
    }
    acc00 = __builtin_amdgcn_wmma_f32_16x16x32_bf16(false, fa0.v, false, fb0.v, (short)0, acc00, false, false);
    acc01 = __builtin_amdgcn_wmma_f32_16x16x32_bf16(false, fa0.v, false, fb1.v, (short)0, acc01, false, false);
    acc10 = __builtin_amdgcn_wmma_f32_16x16x32_bf16(false, fa1.v, false, fb0.v, (short)0, acc10, false, false);
    acc11 = __builtin_amdgcn_wmma_f32_16x16x32_bf16(false, fa1.v, false, fb1.v, (short)0, acc11, false, false);
  }

  // ---- epilogue ----
  #pragma unroll
  for (int mi = 0; mi < 2; ++mi) {
    #pragma unroll
    for (int ni = 0; ni < 2; ++ni) {
      v8f acc = (mi == 0) ? (ni == 0 ? acc00 : acc01) : (ni == 0 ? acc10 : acc11);
      const int col = n0 + wc * 32 + ni * 16 + nlane;
      const float bcol = bias ? bias[col] : 0.0f;
      #pragma unroll
      for (int r = 0; r < 8; ++r) {
        const int row = m0 + wr * 32 + mi * 16 + r + ((lane >> 4) << 3);
        const size_t idx = (size_t)row * Nn + col;
        float v = acc[r];
        if (EPI == 0) {
          outB[idx] = (bf16)(v + bcol);
        } else if (EPI == 1) {
          outF[idx] = resid[idx] + v + bcol;
        } else if (EPI == 2) {
          float xg = v + bcol;
          float gl = 0.5f * xg * (1.0f + erff(xg * 0.70710678118654752f));
          outB[idx] = (bf16)gl;
        } else { // EPI == 3
          float wgt = cw[(size_t)row * EE];
          outF[idx] += wgt * (v + bcol);
        }
      }
    }
  }
}

// ---------------------------------------------------------------------------
// Flash attention, causal, dh=64, bf16 WMMA, f32 online softmax.
// One wave per (b, h, 16-query tile). Key chunks of 32.
// ---------------------------------------------------------------------------
__global__ __launch_bounds__(32)
void attn_kernel(const bf16* __restrict__ Q, const bf16* __restrict__ Kp,
                 const bf16* __restrict__ V, bf16* __restrict__ Y)
{
  const int nQT = TT / 16;
  const int blk = blockIdx.x;
  const int qt = blk % nQT;
  const int bh = blk / nQT;
  const int h = bh % HH;
  const int b = bh / HH;
  const int lane = threadIdx.x;
  const int q0 = qt * 16;
  const size_t base = ((size_t)b * TT) * DD + (size_t)h * DH;

  const int mfrag = lane & 15;
  const int kb = (lane >> 4) << 3;

  FragA qf[2];
  {
    const bf16* qrow = Q + base + (size_t)(q0 + mfrag) * DD;
    #pragma unroll
    for (int f = 0; f < 2; ++f)
      #pragma unroll
      for (int p = 0; p < 4; ++p) {
        qf[f].u[p]     = *(const unsigned*)&qrow[f * 32 + kb + 2 * p];
        qf[f].u[4 + p] = *(const unsigned*)&qrow[f * 32 + 16 + kb + 2 * p];
      }
  }

  float mrow[8], lrow[8];
  v8f o[4];
  #pragma unroll
  for (int r = 0; r < 8; ++r) { mrow[r] = -INFINITY; lrow[r] = 0.f; }
  #pragma unroll
  for (int tt = 0; tt < 4; ++tt)
    #pragma unroll
    for (int r = 0; r < 8; ++r) o[tt][r] = 0.f;

  __shared__ __align__(16) bf16 sP[16 * 32];
  __shared__ __align__(16) bf16 sVT[64 * 32];

  const int ncMax = TT >> 5;
  int nc = (q0 + 47) >> 5; if (nc > ncMax) nc = ncMax;
  const float scl = 0.125f;   // 1/sqrt(64)

  for (int c = 0; c < nc; ++c) {
    const int kc = c << 5;
    v8f s0, s1;
    #pragma unroll
    for (int r = 0; r < 8; ++r) { s0[r] = 0.f; s1[r] = 0.f; }

    FragA kf;
    #pragma unroll
    for (int f = 0; f < 2; ++f) {
      const bf16* kr = Kp + base + (size_t)(kc + mfrag) * DD + f * 32;
      #pragma unroll
      for (int p = 0; p < 4; ++p) {
        kf.u[p]     = *(const unsigned*)&kr[kb + 2 * p];
        kf.u[4 + p] = *(const unsigned*)&kr[16 + kb + 2 * p];
      }
      s0 = __builtin_amdgcn_wmma_f32_16x16x32_bf16(false, qf[f].v, false, kf.v, (short)0, s0, false, false);
    }
    #pragma unroll
    for (int f = 0; f < 2; ++f) {
      const bf16* kr = Kp + base + (size_t)(kc + 16 + mfrag) * DD + f * 32;
      #pragma unroll
      for (int p = 0; p < 4; ++p) {
        kf.u[p]     = *(const unsigned*)&kr[kb + 2 * p];
        kf.u[4 + p] = *(const unsigned*)&kr[16 + kb + 2 * p];
      }
      s1 = __builtin_amdgcn_wmma_f32_16x16x32_bf16(false, qf[f].v, false, kf.v, (short)0, s1, false, false);
    }

    #pragma unroll
    for (int r = 0; r < 8; ++r) {
      const int qrow = q0 + r + ((lane >> 4) << 3);
      float a = s0[r] * scl, bb = s1[r] * scl;
      if (kc + mfrag > qrow) a = -INFINITY;
      if (kc + 16 + mfrag > qrow) bb = -INFINITY;
      float mx = fmaxf(a, bb);
      #pragma unroll
      for (int d = 1; d < 16; d <<= 1) mx = fmaxf(mx, __shfl_xor(mx, d, 32));
      const float nm = fmaxf(mrow[r], mx);
      const float p0 = __expf(a - nm), p1 = __expf(bb - nm);
      float rsum = p0 + p1;
      #pragma unroll
      for (int d = 1; d < 16; d <<= 1) rsum += __shfl_xor(rsum, d, 32);
      const float resc = __expf(mrow[r] - nm);
      lrow[r] = lrow[r] * resc + rsum;
      mrow[r] = nm;
      #pragma unroll
      for (int tt = 0; tt < 4; ++tt) o[tt][r] *= resc;
      s0[r] = p0; s1[r] = p1;
    }

    __syncthreads();
    #pragma unroll
    for (int r = 0; r < 8; ++r) {
      const int row = r + ((lane >> 4) << 3);
      sP[row * 32 + mfrag] = (bf16)s0[r];
      sP[row * 32 + 16 + mfrag] = (bf16)s1[r];
    }
    {
      const bf16* vr = V + base + (size_t)(kc + lane) * DD;
      #pragma unroll
      for (int cp = 0; cp < 32; ++cp) {
        PairU pv; pv.u = *(const unsigned*)&vr[2 * cp];
        sVT[(2 * cp) * 32 + lane] = pv.h[0];
        sVT[(2 * cp + 1) * 32 + lane] = pv.h[1];
      }
    }
    __syncthreads();

    FragA pf;
    #pragma unroll
    for (int p = 0; p < 4; ++p) {
      pf.u[p]     = *(const unsigned*)&sP[mfrag * 32 + kb + 2 * p];
      pf.u[4 + p] = *(const unsigned*)&sP[mfrag * 32 + 16 + kb + 2 * p];
    }
    #pragma unroll
    for (int tt = 0; tt < 4; ++tt) {
      FragA vf;
      const bf16* vp = &sVT[(tt * 16 + mfrag) * 32];
      #pragma unroll
      for (int p = 0; p < 4; ++p) {
        vf.u[p]     = *(const unsigned*)&vp[kb + 2 * p];
        vf.u[4 + p] = *(const unsigned*)&vp[16 + kb + 2 * p];
      }
      o[tt] = __builtin_amdgcn_wmma_f32_16x16x32_bf16(false, pf.v, false, vf.v, (short)0, o[tt], false, false);
    }
  }

  #pragma unroll
  for (int tt = 0; tt < 4; ++tt)
    #pragma unroll
    for (int r = 0; r < 8; ++r) {
      const int row = q0 + r + ((lane >> 4) << 3);
      const int ch = tt * 16 + mfrag;
      Y[base + (size_t)row * DD + ch] = (bf16)(o[tt][r] / lrow[r]);
    }
}

// ---------------------------------------------------------------------------
// Router: gate/noise logits, noisy top-2 softmax, combine weights.
// ---------------------------------------------------------------------------
__global__ __launch_bounds__(32)
void router_kernel(const bf16* __restrict__ H2, const float* __restrict__ wg,
                   const float* __restrict__ bg, const float* __restrict__ wn,
                   const float* __restrict__ bn, const float* __restrict__ noise,
                   float* __restrict__ gate_out, float* __restrict__ noisy_out,
                   float* __restrict__ cw)
{
  const int n = blockIdx.x;
  const int lane = threadIdx.x;
  float ag[EE] = {0.f}, an[EE] = {0.f};
  for (int d = lane; d < DD; d += 32) {
    const float hv = (float)H2[(size_t)n * DD + d];
    #pragma unroll
    for (int e = 0; e < EE; ++e) {
      ag[e] += hv * wg[d * EE + e];
      an[e] += hv * wn[d * EE + e];
    }
  }
  #pragma unroll
  for (int e = 0; e < EE; ++e)
    #pragma unroll
    for (int d = 1; d < 32; d <<= 1) {
      ag[e] += __shfl_xor(ag[e], d, 32);
      an[e] += __shfl_xor(an[e], d, 32);
    }
  if (lane == 0) {
    float nz[EE], rw[EE];
    #pragma unroll
    for (int e = 0; e < EE; ++e) {
      const float gl = ag[e] + bg[e];
      const float nl = an[e] + bn[e];
      const float sp = (nl > 20.f) ? nl : log1pf(__expf(nl));
      nz[e] = gl + noise[(size_t)n * EE + e] * sp;
      gate_out[(size_t)n * EE + e] = gl;
      noisy_out[(size_t)n * EE + e] = nz[e];
    }
    float mx = nz[0];
    #pragma unroll
    for (int e = 1; e < EE; ++e) mx = fmaxf(mx, nz[e]);
    float sum = 0.f;
    #pragma unroll
    for (int e = 0; e < EE; ++e) { rw[e] = __expf(nz[e] - mx); sum += rw[e]; }
    #pragma unroll
    for (int e = 0; e < EE; ++e) rw[e] /= sum;
    int i0 = 0;
    #pragma unroll
    for (int e = 1; e < EE; ++e) if (rw[e] > rw[i0]) i0 = e;
    int i1 = (i0 == 0) ? 1 : 0;
    #pragma unroll
    for (int e = 0; e < EE; ++e) if (e != i0 && rw[e] > rw[i1]) i1 = e;
    const float ts = rw[i0] + rw[i1];
    #pragma unroll
    for (int e = 0; e < EE; ++e) cw[(size_t)n * EE + e] = 0.f;
    cw[(size_t)n * EE + i0] = rw[i0] / ts;
    cw[(size_t)n * EE + i1] = rw[i1] / ts;
  }
}

__global__ __launch_bounds__(256)
void copy_f32x4_kernel(const float4* __restrict__ src, float4* __restrict__ dst, int n4)
{
  const int i = blockIdx.x * blockDim.x + threadIdx.x;
  if (i < n4) dst[i] = src[i];
}

// ---------------------------------------------------------------------------
extern "C" void kernel_launch(void* const* d_in, const int* in_sizes, int n_in,
                              void* d_out, int out_size, void* d_ws, size_t ws_size,
                              hipStream_t stream)
{
  const float* x      = (const float*)d_in[0];
  const float* noise  = (const float*)d_in[1];
  const float* ln1_g  = (const float*)d_in[2];
  const float* ln1_b  = (const float*)d_in[3];
  const float* ln2_g  = (const float*)d_in[4];
  const float* ln2_b  = (const float*)d_in[5];
  const float* wq = (const float*)d_in[6];  const float* bq = (const float*)d_in[7];
  const float* wk = (const float*)d_in[8];  const float* bk = (const float*)d_in[9];
  const float* wv = (const float*)d_in[10]; const float* bv = (const float*)d_in[11];
  const float* wp = (const float*)d_in[12]; const float* bp = (const float*)d_in[13];
  const float* wg = (const float*)d_in[14]; const float* bg = (const float*)d_in[15];
  const float* wn = (const float*)d_in[16]; const float* bn = (const float*)d_in[17];
  const float* w_in  = (const float*)d_in[18]; const float* b_in  = (const float*)d_in[19];
  const float* w_out = (const float*)d_in[20]; const float* b_out = (const float*)d_in[21];

  float* out_x     = (float*)d_out;                         // [N, D]
  float* out_noisy = out_x + (size_t)NTOK * DD;             // [N, E]
  float* out_gate  = out_noisy + (size_t)NTOK * EE;         // [N, E]

  char* ws = (char*)d_ws;
  size_t off = 0;
  auto take = [&](size_t bytes) { void* p = ws + off; off += (bytes + 255) & ~(size_t)255; return p; };
  bf16* h1  = (bf16*)take((size_t)NTOK * DD * sizeof(bf16));
  bf16* qb  = (bf16*)take((size_t)NTOK * DD * sizeof(bf16));
  bf16* kb  = (bf16*)take((size_t)NTOK * DD * sizeof(bf16));
  bf16* vb  = (bf16*)take((size_t)NTOK * DD * sizeof(bf16));
  bf16* yb  = (bf16*)take((size_t)NTOK * DD * sizeof(bf16));
  float* x1 = (float*)take((size_t)NTOK * DD * sizeof(float));
  bf16* h2  = (bf16*)take((size_t)NTOK * DD * sizeof(bf16));
  float* cw = (float*)take((size_t)NTOK * EE * sizeof(float));
  bf16* mid = (bf16*)take((size_t)NTOK * FF * sizeof(bf16));
  (void)ws_size;

  // 1. LN1
  ln_kernel<<<NTOK, 256, 0, stream>>>(x, ln1_g, ln1_b, h1);

  // 2. q/k/v projections (bf16 out)
  dim3 gQKV(DD / 128, NTOK / 64);
  gemm_bf16_kernel<0><<<gQKV, 256, 0, stream>>>(h1, wq, bq, nullptr, nullptr, qb, nullptr, NTOK, DD, DD);
  gemm_bf16_kernel<0><<<gQKV, 256, 0, stream>>>(h1, wk, bk, nullptr, nullptr, kb, nullptr, NTOK, DD, DD);
  gemm_bf16_kernel<0><<<gQKV, 256, 0, stream>>>(h1, wv, bv, nullptr, nullptr, vb, nullptr, NTOK, DD, DD);

  // 3. causal flash attention
  attn_kernel<<<NB * HH * (TT / 16), 32, 0, stream>>>(qb, kb, vb, yb);

  // 4. x1 = x + y @ wp + bp   (fp32)
  gemm_bf16_kernel<1><<<gQKV, 256, 0, stream>>>(yb, wp, bp, x, nullptr, nullptr, x1, NTOK, DD, DD);

  // 5. LN2
  ln_kernel<<<NTOK, 256, 0, stream>>>(x1, ln2_g, ln2_b, h2);

  // 6. router
  router_kernel<<<NTOK, 32, 0, stream>>>(h2, wg, bg, wn, bn, noise, out_gate, out_noisy, cw);

  // 7. out = x (original residual)
  {
    const int n4 = NTOK * DD / 4;
    copy_f32x4_kernel<<<(n4 + 255) / 256, 256, 0, stream>>>((const float4*)x, (float4*)out_x, n4);
  }

  // 8. routed experts
  dim3 gIN(FF / 128, NTOK / 64);
  dim3 gOUT(DD / 128, NTOK / 64);
  for (int e = 0; e < EE; ++e) {
    gemm_bf16_kernel<2><<<gIN, 256, 0, stream>>>(h2, w_in + (size_t)e * DD * FF, b_in + (size_t)e * FF,
                                                 nullptr, cw + e, mid, nullptr, NTOK, FF, DD);
    gemm_bf16_kernel<3><<<gOUT, 256, 0, stream>>>(mid, w_out + (size_t)e * FF * DD, b_out + (size_t)e * DD,
                                                  nullptr, cw + e, nullptr, out_x, NTOK, DD, FF);
  }
}
